// SparseUNetEncoder_18279380811798
// MI455X (gfx1250) — matrix operand order repaired and checked by reading, not compiled, not run on hardware
//
#include <hip/hip_runtime.h>

// ---------------------------------------------------------------------------
// CDNA5 (gfx1250, wave32) sparse UNet encoder.
// Core op: gather-GEMM via V_WMMA_F32_16X16X32_BF16.
//  - 128-thread block = 4 waves; block tile = 64 rows x 64 couts.
//  - Each wave owns 16 rows and FOUR 16x16 accumulators (4 cout tiles), so one
//    gathered A-fragment feeds 4 WMMAs (4x less feature gather traffic).
//  - Weight slab [64 x CinPad] bf16 staged to LDS per kernel-offset k via the
//    Tensor Data Mover (TENSOR_LOAD_TO_LDS + s_wait_tensorcnt); B-fragments
//    read back as ds_load_b128 pairs.
//  - Neighbor idx/mask broadcast across lane halves via ds_bpermute (__shfl).
//  - CinPad is a template parameter so the WMMA chains fully unroll.
// ---------------------------------------------------------------------------

typedef __attribute__((ext_vector_type(16))) __bf16   v16bf;
typedef __attribute__((ext_vector_type(8)))  float    v8f;
typedef __attribute__((ext_vector_type(4)))  unsigned v4u;
typedef __attribute__((ext_vector_type(4)))  int      v4i;
typedef __attribute__((ext_vector_type(8)))  int      v8i;

#if defined(__has_builtin)
#if __has_builtin(__builtin_amdgcn_tensor_load_to_lds) && \
    __has_builtin(__builtin_amdgcn_s_wait_tensorcnt)
#define HAVE_TDM 1
#endif
#endif

// ---- weight convert: fp32 [K][Cin][Cout] -> bf16 [K][Cout][CinPad] ---------
__global__ __launch_bounds__(256)
void wconv_kernel(const float* __restrict__ W, __bf16* __restrict__ Wt,
                  int K, int Cin, int CinPad, int Cout) {
  int idx = blockIdx.x * blockDim.x + threadIdx.x;
  int total = K * Cout * CinPad;
  if (idx >= total) return;
  int cin = idx % CinPad;
  int t   = idx / CinPad;
  int co  = t % Cout;
  int k   = t / Cout;
  float v = (cin < Cin) ? W[((size_t)k * Cin + cin) * Cout + co] : 0.0f;
  Wt[idx] = (__bf16)v;
}

// ---- TDM stage: contiguous slab of `elems` bf16 from global -> LDS ---------
__device__ __forceinline__ void tdm_stage_bf16(const __bf16* gsrc, __bf16* lds,
                                               unsigned elems) {
#if defined(HAVE_TDM)
  unsigned long long ga = (unsigned long long)gsrc;
  // LDS aperture: addr[31:0] is the in-allocation LDS byte offset.
  unsigned lds_off = (unsigned)(unsigned long long)(void*)lds;
  v4u g0;
  g0[0] = 1u;                                   // count=1, user-mode D#
  g0[1] = lds_off;                              // lds_addr
  g0[2] = (unsigned)ga;                         // global_addr[31:0]
  g0[3] = (unsigned)(ga >> 32) | (2u << 30);    // global_addr[56:32] | type=2
  v8i g1;
  g1[0] = (int)(1u << 16);                      // wg_mask=0, data_size=1 (2B)
  g1[1] = (int)((elems & 0xFFFFu) << 16);       // tensor_dim0[15:0]
  g1[2] = (int)(((elems >> 16) & 0xFFFFu) | (1u << 16));  // dim0 hi | tensor_dim1=1
  g1[3] = (int)((elems & 0xFFFFu) << 16);       // tensor_dim1 hi=0 | tile_dim0
  g1[4] = (int)1;                               // tile_dim1=1, tile_dim2=0
  g1[5] = (int)elems;                           // tensor_dim0_stride lo
  g1[6] = 0;
  g1[7] = 0;
  v4i z4 = {};
#if defined(__clang_major__) && (__clang_major__ >= 23)
  v8i z8 = {};
  __builtin_amdgcn_tensor_load_to_lds(g0, g1, z4, z4, z8, 0);
#else
  __builtin_amdgcn_tensor_load_to_lds(g0, g1, z4, z4, 0);
#endif
  __builtin_amdgcn_s_wait_tensorcnt(0);
#else
  // cooperative fallback (all 128 threads of the block enter here)
  const uint4* src = (const uint4*)gsrc;
  uint4* dst = (uint4*)lds;
  const int units = (int)(elems * 2u / 16u);
  for (int i = threadIdx.x; i < units; i += 128) dst[i] = src[i];
#endif
}

// ---- sparse conv: out[Nout,Cout] = sum_k (feats[nbr_k]*mask_k) @ W_k -------
// grid = (ceil(Nout/64), ceil(Cout/64)), block = 128 (4 waves x 16-row tiles),
// each wave accumulates 4 cout tiles. nbr==nullptr => identity map (1x1 proj).
template <int CINPAD>
__global__ __launch_bounds__(128)
void spconv_wmma_kernel(const float* __restrict__ feats,
                        const __bf16* __restrict__ Wt,   // [K][Cout][CINPAD]
                        const int*   __restrict__ nbr,   // [K][Nout] or null
                        const float* __restrict__ mask,  // [K][Nout] or null
                        float* __restrict__ out,         // [Nout][Cout]
                        int Nout, int Cin, int Cout, int K) {
  __shared__ __bf16 wsm[64 * CINPAD];          // weight slab for current k

  const int tid  = threadIdx.x;
  const int lane = tid & 31;
  const int wave = tid >> 5;
  const int m    = lane & 15;                  // row within tile / B column
  const int kh   = lane >> 4;                  // K-half selector
  const int rowBase  = blockIdx.x * 64 + wave * 16;
  const int coutBase = blockIdx.y * 64;
  const int row = rowBase + m;

  v8f acc[4] = {v8f{}, v8f{}, v8f{}, v8f{}};

  for (int k = 0; k < K; ++k) {
    const __bf16* wk = Wt + ((size_t)k * Cout + coutBase) * CINPAD;
    __syncthreads();                            // prior reads done before overwrite
#if defined(HAVE_TDM)
    if (wave == 0) tdm_stage_bf16(wk, wsm, 64u * CINPAD);
#else
    tdm_stage_bf16(wk, wsm, 64u * CINPAD);
#endif
    __syncthreads();                            // slab visible to all 4 waves

    // neighbor index + mask for this wave's 16 rows, broadcast to both halves
    int nidx; float mk;
    if (nbr) {
      int ti = 0; float tf = 0.0f;
      if (lane < 16) {
        int r = rowBase + lane;
        if (r < Nout) {
          ti = nbr [(size_t)k * Nout + r];
          tf = mask[(size_t)k * Nout + r];
        }
      }
      nidx = __shfl(ti, m, 32);
      mk   = __shfl(tf, m, 32);
    } else {
      nidx = (row < Nout) ? row : 0;
      mk   = (row < Nout) ? 1.0f : 0.0f;
    }
    const float* fr = feats + (size_t)nidx * Cin;

#pragma unroll
    for (int cc = 0; cc < CINPAD / 32; ++cc) {
      const int cb = cc << 5;

      // A fragment: lane holds row m; elems 0..7 <-> K=kh*8+j, 8..15 <-> 16+kh*8+j
      v16bf a;
      if (cb + 32 <= Cin) {
        const float4* pa = (const float4*)(fr + cb + kh * 8);
        const float4* pb = (const float4*)(fr + cb + 16 + kh * 8);
        float4 a0 = pa[0], a1 = pa[1];
        float4 b0 = pb[0], b1 = pb[1];
        a[0]  = (__bf16)(a0.x * mk); a[1]  = (__bf16)(a0.y * mk);
        a[2]  = (__bf16)(a0.z * mk); a[3]  = (__bf16)(a0.w * mk);
        a[4]  = (__bf16)(a1.x * mk); a[5]  = (__bf16)(a1.y * mk);
        a[6]  = (__bf16)(a1.z * mk); a[7]  = (__bf16)(a1.w * mk);
        a[8]  = (__bf16)(b0.x * mk); a[9]  = (__bf16)(b0.y * mk);
        a[10] = (__bf16)(b0.z * mk); a[11] = (__bf16)(b0.w * mk);
        a[12] = (__bf16)(b1.x * mk); a[13] = (__bf16)(b1.y * mk);
        a[14] = (__bf16)(b1.z * mk); a[15] = (__bf16)(b1.w * mk);
      } else {  // padded tail (only the Cin==8 first conv)
#pragma unroll
        for (int j = 0; j < 8; ++j) {
          int c0 = cb + kh * 8 + j;
          int c1 = cb + 16 + kh * 8 + j;
          a[j]     = (__bf16)((c0 < Cin) ? fr[c0] * mk : 0.0f);
          a[j + 8] = (__bf16)((c1 < Cin) ? fr[c1] * mk : 0.0f);
        }
      }

      // 4 cout tiles share this A fragment; B from LDS (two 8-bf16 halves: K=kh*8+j, 16+kh*8+j)
#pragma unroll
      for (int nt = 0; nt < 4; ++nt) {
        typedef __attribute__((ext_vector_type(8))) __bf16 v8bf_t;
        const __bf16* wb = wsm + (nt * 16 + m) * CINPAD + cb + kh * 8;
        v8bf_t blo = *(const v8bf_t*)(wb);
        v8bf_t bhi = *(const v8bf_t*)(wb + 16);
        v16bf b = __builtin_shufflevector(blo, bhi, 0,1,2,3,4,5,6,7,8,9,10,11,12,13,14,15);
        acc[nt] = __builtin_amdgcn_wmma_f32_16x16x32_bf16(
            false, a, false, b, (short)0, acc[nt], false, false);
      }
    }
  }

  // C/D layout: VGPR r -> M = r + kh*8, N = m (per cout tile)
#pragma unroll
  for (int nt = 0; nt < 4; ++nt) {
    int cb = coutBase + nt * 16;
    if (cb < Cout) {
#pragma unroll
      for (int r = 0; r < 8; ++r) {
        int orow = rowBase + kh * 8 + r;
        if (orow < Nout) out[(size_t)orow * Cout + cb + m] = acc[nt][r];
      }
    }
  }
}

// ---- BatchNorm column stats: stats[0..C) = sum, stats[C..2C) = sumsq -------
__global__ void bn_stats_kernel(const float* __restrict__ x, int N, int C,
                                int rowsPerBlock, float* __restrict__ stats) {
  int c  = threadIdx.x;                // blockDim.x == C (32..512)
  int r0 = blockIdx.x * rowsPerBlock;
  int r1 = r0 + rowsPerBlock; if (r1 > N) r1 = N;
  float s = 0.0f, ss = 0.0f;
  for (int r = r0; r < r1; ++r) {
    float v = x[(size_t)r * C + c];
    s += v; ss += v * v;
  }
  atomicAdd(&stats[c], s);
  atomicAdd(&stats[C + c], ss);
}

// ---- BN apply (+ optional residual, optional ReLU) -------------------------
__global__ __launch_bounds__(256)
void bn_apply_kernel(const float* __restrict__ x, const float* __restrict__ stats,
                     const float* __restrict__ gamma, const float* __restrict__ beta,
                     const float* __restrict__ residual, float* __restrict__ out,
                     int N, int C, int relu) {
  size_t i = (size_t)blockIdx.x * blockDim.x + threadIdx.x;
  size_t tot = (size_t)N * C;
  if (i >= tot) return;
  int c = (int)(i % C);
  float invN = 1.0f / (float)N;
  float mn = stats[c] * invN;
  float vr = stats[C + c] * invN - mn * mn;
  float y = (x[i] - mn) * rsqrtf(vr + 1e-5f) * gamma[c] + beta[c];
  if (residual) y += residual[i];
  if (relu) y = fmaxf(y, 0.0f);
  out[i] = y;
}

// ---------------------------------------------------------------------------
// Host orchestration. Input flatten order (jax tree: dict keys sorted):
//   0: feats
//   1-4:  maps.dn_mask[0..3]   5-8:  maps.dn_nbr[0..3]
//   9-13: maps.mask[0..4]      14-18: maps.nbr[0..4]
//   19-24: bottleneck rb0 (W1,W2,b1,b2,g1,g2)   25-30: bottleneck rb1
//   31+16*l: level l -> down(W,b,g); rb0(W1,W2,Wp,b1,b2,g1,g2); rb1(W1,W2,b1,b2,g1,g2)
// ---------------------------------------------------------------------------
extern "C" void kernel_launch(void* const* d_in, const int* in_sizes, int n_in,
                              void* d_out_v, int out_size, void* d_ws, size_t ws_size,
                              hipStream_t stream) {
  (void)n_in; (void)out_size; (void)ws_size;

  const float* feats = (const float*)d_in[0];
  const float* dn_mask[4]; const int* dn_nbr[4];
  const float* mask[5];    const int* nbr[5];
  for (int l = 0; l < 4; ++l) {
    dn_mask[l] = (const float*)d_in[1 + l];
    dn_nbr[l]  = (const int*)  d_in[5 + l];
  }
  for (int l = 0; l < 5; ++l) {
    mask[l] = (const float*)d_in[9 + l];
    nbr[l]  = (const int*)  d_in[14 + l];
  }

  int N[5];
  for (int l = 0; l < 5; ++l) N[l] = in_sizes[9 + l] / 27;
  const int CH[5] = {32, 64, 128, 256, 512};

  float* dout = (float*)d_out_v;
  size_t outOff[5]; outOff[0] = 0;
  for (int l = 1; l < 5; ++l) outOff[l] = outOff[l - 1] + (size_t)N[l - 1] * CH[l - 1];

  // ---- workspace carve-out
  char* ws = (char*)d_ws; size_t off = 0;
  auto alloc = [&](size_t bytes) -> void* {
    off = (off + 255) & ~(size_t)255;
    void* p = ws + off; off += bytes; return p;
  };
  size_t maxAct = 0;
  for (int l = 0; l < 5; ++l) {
    size_t a = (size_t)N[l] * CH[l];
    if (a > maxAct) maxAct = a;
  }
  __bf16* wbuf  = (__bf16*)alloc((size_t)27 * 512 * 512 * sizeof(__bf16));
  float*  stats = (float*) alloc(2 * 512 * sizeof(float));
  float*  convT = (float*) alloc(maxAct * sizeof(float));
  float*  projT = (float*) alloc(maxAct * sizeof(float));
  float*  bufH  = (float*) alloc(maxAct * sizeof(float));
  float*  bufA  = (float*) alloc(maxAct * sizeof(float));
  float*  bufB  = (float*) alloc(maxAct * sizeof(float));

  auto F = [&](int i) { return (const float*)d_in[i]; };

  auto conv = [&](const float* x, int Cin, const float* W, int K,
                  const int* nb, const float* mk, int Nout, int Cout, float* out) {
    int CinPad = (Cin + 31) & ~31;
    int tot = K * Cout * CinPad;
    wconv_kernel<<<dim3((tot + 255) / 256), dim3(256), 0, stream>>>(
        W, wbuf, K, Cin, CinPad, Cout);
    dim3 g((Nout + 63) / 64, (Cout + 63) / 64);
    switch (CinPad) {
      case 32:
        spconv_wmma_kernel<32><<<g, dim3(128), 0, stream>>>(
            x, wbuf, nb, mk, out, Nout, Cin, Cout, K); break;
      case 64:
        spconv_wmma_kernel<64><<<g, dim3(128), 0, stream>>>(
            x, wbuf, nb, mk, out, Nout, Cin, Cout, K); break;
      case 128:
        spconv_wmma_kernel<128><<<g, dim3(128), 0, stream>>>(
            x, wbuf, nb, mk, out, Nout, Cin, Cout, K); break;
      case 256:
        spconv_wmma_kernel<256><<<g, dim3(128), 0, stream>>>(
            x, wbuf, nb, mk, out, Nout, Cin, Cout, K); break;
      default:
        spconv_wmma_kernel<512><<<g, dim3(128), 0, stream>>>(
            x, wbuf, nb, mk, out, Nout, Cin, Cout, K); break;
    }
  };

  auto bn = [&](const float* x, int Nr, int C, const float* g_, const float* b_,
                const float* res, int relu, float* out) {
    hipMemsetAsync(stats, 0, 2 * (size_t)C * sizeof(float), stream);
    const int ROWS = 512;
    bn_stats_kernel<<<dim3((Nr + ROWS - 1) / ROWS), dim3(C), 0, stream>>>(
        x, Nr, C, ROWS, stats);
    size_t tot = (size_t)Nr * C;
    bn_apply_kernel<<<dim3((unsigned)((tot + 255) / 256)), dim3(256), 0, stream>>>(
        x, stats, g_, b_, res, out, Nr, C, relu);
  };

  struct RB { const float *W1, *W2, *Wp, *g1, *b1, *g2, *b2; };
  auto resblock = [&](const float* x, int Nl, int Cin, int Cout,
                      const int* nb, const float* mk, const RB& p, float* out) {
    conv(x, Cin, p.W1, 27, nb, mk, Nl, Cout, convT);
    bn(convT, Nl, Cout, p.g1, p.b1, nullptr, 1, bufH);
    conv(bufH, Cout, p.W2, 27, nb, mk, Nl, Cout, convT);
    const float* idn = x;
    if (p.Wp) { conv(x, Cin, p.Wp, 1, nullptr, nullptr, Nl, Cout, projT); idn = projT; }
    bn(convT, Nl, Cout, p.g2, p.b2, idn, 1, out);
  };

  const float* cur = feats;
  int Ccur = 8;
  float* pp[2] = {bufA, bufB};
  int ping = 0;

  for (int l = 0; l < 4; ++l) {
    const int B  = 31 + 16 * l;
    const int Cl = CH[l];
    RB rb0{F(B + 3), F(B + 4), F(B + 5), F(B + 8), F(B + 6), F(B + 9), F(B + 7)};
    RB rb1{F(B + 10), F(B + 11), nullptr, F(B + 14), F(B + 12), F(B + 15), F(B + 13)};

    float* o1 = pp[ping];
    resblock(cur, N[l], Ccur, Cl, nbr[l], mask[l], rb0, o1);
    float* lvlOut = dout + outOff[l];
    resblock(o1, N[l], Cl, Cl, nbr[l], mask[l], rb1, lvlOut);

    int Cnext = (l == 3) ? 512 : CH[l + 1];
    conv(lvlOut, Cl, F(B), 8, dn_nbr[l], dn_mask[l], N[l + 1], Cnext, convT);
    bn(convT, N[l + 1], Cnext, F(B + 2), F(B + 1), nullptr, 1, pp[ping ^ 1]);
    cur = pp[ping ^ 1]; Ccur = Cnext; ping ^= 1;
  }

  RB bb0{F(19), F(20), nullptr, F(23), F(21), F(24), F(22)};
  RB bb1{F(25), F(26), nullptr, F(29), F(27), F(30), F(28)};
  float* o = pp[ping];
  resblock(cur, N[4], 512, 512, nbr[4], mask[4], bb0, o);
  resblock(o,   N[4], 512, 512, nbr[4], mask[4], bb1, dout + outOff[4]);
}
